// SaliencyExtractor_26594437497194
// MI455X (gfx1250) — matrix operand order repaired and checked by reading, not compiled, not run on hardware
//
#include <hip/hip_runtime.h>
#include <hip/hip_bf16.h>

typedef __attribute__((ext_vector_type(16))) _Float16 v16h;
typedef __attribute__((ext_vector_type(8)))  _Float16 v8h;
typedef __attribute__((ext_vector_type(8)))  float    v8f;

#define BATCH 16
#define NPTS  1024
#define HH    224
#define WW    224
#define KSZ   23
#define HALF  11
#define TILES_N 14                 // 224 / 16
#define NTILES  (TILES_N * TILES_N) // 196
#define WAVES_PER_BLOCK 4
#define KTILE 32
#define KITERS (WW / KTILE)        // 7

union V16U { v16h v; _Float16 e[16]; };
union V8U  { v8h  v; _Float16 e[8];  };

// ---------------------------------------------------------------------------
// 1) Zero the impulse maps and build the symmetric banded Gaussian matrix K
//    K[r,c] = k1[c-r+11] for |c-r|<=11, else 0.   (f16)
// ---------------------------------------------------------------------------
__global__ void sal_init_kernel(float* __restrict__ imp, _Float16* __restrict__ kmat) {
    int i = blockIdx.x * blockDim.x + threadIdx.x;
    if (i < BATCH * HH * WW) imp[i] = 0.0f;
    if (i < HH * WW) {
        int r = i / WW, c = i % WW;
        int d = c - r;
        float w = 0.0f;
        if (d >= -HALF && d <= HALF) {
            float norm = 0.0f;
            #pragma unroll
            for (int t = -HALF; t <= HALF; ++t)
                norm += __expf((float)(-t * t) * (1.0f / 18.0f));
            w = __expf((float)(-d * d) * (1.0f / 18.0f)) / norm;
        }
        kmat[i] = (_Float16)w;
    }
}

// ---------------------------------------------------------------------------
// 2) Scatter point impulses: imp[b, floor(py*H), floor(px*W)] += 1
// ---------------------------------------------------------------------------
__global__ void sal_scatter_kernel(const float* __restrict__ pts, float* __restrict__ imp) {
    int i = blockIdx.x * blockDim.x + threadIdx.x;
    if (i >= BATCH * NPTS) return;
    int b = i / NPTS;
    float px = pts[2 * i + 0];
    float py = pts[2 * i + 1];
    int x = (int)(px * (float)WW);
    int y = (int)(py * (float)HH);
    x = min(max(x, 0), WW - 1);
    y = min(max(y, 0), HH - 1);
    atomicAdd(&imp[((size_t)b * HH + y) * WW + x], 1.0f);
}

// ---------------------------------------------------------------------------
// 3) GEMM1: S1 = Imp * K  (row convolution), store S1^T as f16 so the second
//    pass reads contiguously. One wave per 16x16 output tile.
//    A-fragment (f16 16x32): lane l, half i -> K = 16*(l>>4) + i
//    B-fragment (f16 32x16): lane l, half i -> K = 16*(l>>4) + i, N = l&15;
//    loaded via symmetry K[k,n] == K[n,k] -> contiguous along k.
// ---------------------------------------------------------------------------
__global__ void sal_gemm1_kernel(const float* __restrict__ imp,
                                 const _Float16* __restrict__ kmat,
                                 _Float16* __restrict__ tmpT) {
    const int wave = threadIdx.x >> 5;
    const int lane = threadIdx.x & 31;
    const int t = blockIdx.x * WAVES_PER_BLOCK + wave;
    if (t >= NTILES) return;                 // wave-uniform: EXEC stays all-ones
    const int b  = blockIdx.y;
    const int m0 = (t / TILES_N) * 16;
    const int n0 = (t % TILES_N) * 16;
    const int lm = lane & 15;
    const int hi = lane >> 4;

    const float*    arow = imp  + ((size_t)b * HH + (m0 + lm)) * WW;
    const _Float16* brow = kmat + (size_t)(n0 + lm) * WW;   // symmetric access

    v8f c = {};
    for (int kt = 0; kt < KITERS; ++kt) {
        const int k0 = kt * KTILE;

        // prefetch next A strip (global_prefetch_b8 path)
        __builtin_prefetch(arow + k0 + KTILE, 0, 3);

        // ---- A: load 2x (8 contiguous f32), convert to f16 fragment ----
        const float* ap = arow + k0 + 16 * hi;
        float4 f0 = *reinterpret_cast<const float4*>(ap);
        float4 f1 = *reinterpret_cast<const float4*>(ap + 4);
        float4 f2 = *reinterpret_cast<const float4*>(ap + 8);
        float4 f3 = *reinterpret_cast<const float4*>(ap + 12);
        V16U au;
        au.e[0]  = (_Float16)f0.x; au.e[1]  = (_Float16)f0.y;
        au.e[2]  = (_Float16)f0.z; au.e[3]  = (_Float16)f0.w;
        au.e[4]  = (_Float16)f1.x; au.e[5]  = (_Float16)f1.y;
        au.e[6]  = (_Float16)f1.z; au.e[7]  = (_Float16)f1.w;
        au.e[8]  = (_Float16)f2.x; au.e[9]  = (_Float16)f2.y;
        au.e[10] = (_Float16)f2.z; au.e[11] = (_Float16)f2.w;
        au.e[12] = (_Float16)f3.x; au.e[13] = (_Float16)f3.y;
        au.e[14] = (_Float16)f3.z; au.e[15] = (_Float16)f3.w;

        // ---- B: 16 contiguous f16 (two 16B loads) ----
        const _Float16* bp = brow + k0 + 16 * hi;
        V16U bu;
        *reinterpret_cast<v8h*>(&bu.e[0]) = *reinterpret_cast<const v8h*>(bp);
        *reinterpret_cast<v8h*>(&bu.e[8]) = *reinterpret_cast<const v8h*>(bp + 8);

        c = __builtin_amdgcn_wmma_f32_16x16x32_f16(
                false, au.v, false, bu.v, (short)0, c, false, false);
    }

    // D layout: VGPR j -> (M = m0 + j + 8*hi, N = n0 + lm). Store transposed:
    // tmpT[n][m] row n0+lm, cols m0+8*hi .. +7 -> one 16B f16 store.
    V8U du;
    du.e[0] = (_Float16)c[0]; du.e[1] = (_Float16)c[1];
    du.e[2] = (_Float16)c[2]; du.e[3] = (_Float16)c[3];
    du.e[4] = (_Float16)c[4]; du.e[5] = (_Float16)c[5];
    du.e[6] = (_Float16)c[6]; du.e[7] = (_Float16)c[7];
    *reinterpret_cast<v8h*>(tmpT + ((size_t)b * HH + (n0 + lm)) * WW + m0 + 8 * hi) = du.v;
}

// ---------------------------------------------------------------------------
// 4) GEMM2: S = K * S1  (column convolution). B tile (k,n) = tmpT[n][k],
//    contiguous along k. D written as f32 directly to the output.
// ---------------------------------------------------------------------------
__global__ void sal_gemm2_kernel(const _Float16* __restrict__ kmat,
                                 const _Float16* __restrict__ tmpT,
                                 float* __restrict__ out) {
    const int wave = threadIdx.x >> 5;
    const int lane = threadIdx.x & 31;
    const int t = blockIdx.x * WAVES_PER_BLOCK + wave;
    if (t >= NTILES) return;
    const int b  = blockIdx.y;
    const int m0 = (t / TILES_N) * 16;
    const int n0 = (t % TILES_N) * 16;
    const int lm = lane & 15;
    const int hi = lane >> 4;

    const _Float16* arow = kmat + (size_t)(m0 + lm) * WW;
    const _Float16* brow = tmpT + ((size_t)b * HH + (n0 + lm)) * WW;

    v8f c = {};
    for (int kt = 0; kt < KITERS; ++kt) {
        const int k0 = kt * KTILE;

        // ---- A: two 8-half contiguous runs ----
        const _Float16* ap = arow + k0 + 16 * hi;
        V16U au;
        *reinterpret_cast<v8h*>(&au.e[0]) = *reinterpret_cast<const v8h*>(ap);
        *reinterpret_cast<v8h*>(&au.e[8]) = *reinterpret_cast<const v8h*>(ap + 8);

        // ---- B: 16 contiguous halves ----
        const _Float16* bp = brow + k0 + 16 * hi;
        V16U bu;
        *reinterpret_cast<v8h*>(&bu.e[0]) = *reinterpret_cast<const v8h*>(bp);
        *reinterpret_cast<v8h*>(&bu.e[8]) = *reinterpret_cast<const v8h*>(bp + 8);

        c = __builtin_amdgcn_wmma_f32_16x16x32_f16(
                false, au.v, false, bu.v, (short)0, c, false, false);
    }

    // D: VGPR j -> (M = m0 + j + 8*hi, N = n0 + lm); f32 stores stride W.
    float* o = out + ((size_t)b * HH + (m0 + 8 * hi)) * WW + (n0 + lm);
    o[0 * WW] = c[0]; o[1 * WW] = c[1]; o[2 * WW] = c[2]; o[3 * WW] = c[3];
    o[4 * WW] = c[4]; o[5 * WW] = c[5]; o[6 * WW] = c[6]; o[7 * WW] = c[7];
}

// ---------------------------------------------------------------------------
// Workspace layout (all 16B aligned):
//   imp  : BATCH*H*W f32   = 3,211,264 B
//   kmat : H*W f16         =   100,352 B
//   tmpT : BATCH*H*W f16   = 1,605,632 B   (total ~4.9 MB)
// ---------------------------------------------------------------------------
extern "C" void kernel_launch(void* const* d_in, const int* in_sizes, int n_in,
                              void* d_out, int out_size, void* d_ws, size_t ws_size,
                              hipStream_t stream) {
    (void)in_sizes; (void)n_in; (void)out_size; (void)ws_size;
    // d_in[0] = feature_map: used only for its SHAPE by the reference -> never read.
    const float* pts = (const float*)d_in[1];
    float* out = (float*)d_out;

    char* ws = (char*)d_ws;
    float*    imp  = (float*)ws;
    _Float16* kmat = (_Float16*)(ws + (size_t)3211264);
    _Float16* tmpT = (_Float16*)(ws + (size_t)3211264 + 100352);

    {
        int n = BATCH * HH * WW;
        sal_init_kernel<<<(n + 255) / 256, 256, 0, stream>>>(imp, kmat);
    }
    {
        int n = BATCH * NPTS;
        sal_scatter_kernel<<<(n + 255) / 256, 256, 0, stream>>>(pts, imp);
    }
    dim3 grid((NTILES + WAVES_PER_BLOCK - 1) / WAVES_PER_BLOCK, BATCH);
    dim3 block(WAVES_PER_BLOCK * 32);
    sal_gemm1_kernel<<<grid, block, 0, stream>>>(imp, kmat, tmpT);
    sal_gemm2_kernel<<<grid, block, 0, stream>>>(kmat, tmpT, out);
}
